// Performer_49349174231209
// MI455X (gfx1250) — matrix operand order repaired and checked by reading, not compiled, not run on hardware
//
#include <hip/hip_runtime.h>
#include <hip/hip_bf16.h>
#include <math.h>

typedef __bf16 bf16;
typedef bf16  v8bf  __attribute__((ext_vector_type(8)));
typedef bf16  v16bf __attribute__((ext_vector_type(16)));
typedef float v8f   __attribute__((ext_vector_type(8)));
typedef float v4f   __attribute__((ext_vector_type(4)));

// Problem constants (fixed by the reference)
constexpr int   Bt = 4, Nt = 4096, Et = 1024, Ht = 16, Dt = 64, Mt = 256;
constexpr int   ROWS = Bt * Nt;        // 16384 token rows
constexpr int   FR   = Bt * Ht * Nt;   // 262144 feature rows (b,h,n)
constexpr float QK_SCALE = 0.35355339059327379f; // 64^-0.25
constexpr float RATIO    = 0.0625f;              // 256^-0.5
constexpr float KEPS     = 1e-4f;

enum { EP_F32 = 0, EP_QKV = 1, EP_BF16 = 2, EP_OUT = 3 };

// CDNA5 async global->LDS copy (16B), GVS addressing: SGPR base + VGPR offset.
// LDS dest VGPR = low 32 bits of the shared-memory pointer (wave-relative DS addr).
#define ASYNC_LDS_B128(ldsaddr, base, byteoff)                                 \
    asm volatile("global_load_async_to_lds_b128 %0, %1, %2"                    \
                 :: "v"(ldsaddr), "v"(byteoff), "s"(base) : "memory")

// ---------------------------------------------------------------------------
// Templated WMMA GEMM: C = A * B^T (+bias) with layout/epilogue variants.
// Block: 256 threads = 8 waves; block tile 128x128; wave tile 64x32 (4x2 WMMA).
// Double-buffered LDS with async-prefetch of the next K-tile.
// Launch contract: grid fully covers Mr (Mr % 128 == 0 per batch), K % 32 == 0,
// Nc % 16 == 0 (column tiles are uniformly in/out of range).
// ---------------------------------------------------------------------------
template<int MODE, bool TA, bool TB>
__global__ __launch_bounds__(256)
void wmma_gemm(const bf16* __restrict__ A, const bf16* __restrict__ B,
               const float* __restrict__ bias, const float* __restrict__ dinv,
               float* __restrict__ Cf, bf16* __restrict__ Cb,
               int Mr, int Nc, int K, int lda, int ldb, int ldc,
               long sA, long sB, float scale)
{
    __shared__ __attribute__((aligned(16))) bf16 As[2][128][48]; // 96B stride, 16B subrows
    __shared__ __attribute__((aligned(16))) bf16 Bs[2][128][48];
    constexpr int PTILE = (!TA ? 2 : 0) + (!TB ? 2 : 0);   // asyncs per tile per thread

    const int tid  = threadIdx.x;
    const int lane = tid & 31;
    const int wid  = tid >> 5;
    const int wM   = wid >> 2;          // 0..1
    const int wN   = wid & 3;           // 0..3
    const int z    = blockIdx.z;
    const bf16* Ab = A + (long)z * sA;
    const bf16* Bb = B + (long)z * sB;
    const unsigned long long abase = (unsigned long long)(size_t)Ab;
    const unsigned long long bbase = (unsigned long long)(size_t)Bb;
    const int m0 = blockIdx.y * 128;
    const int n0 = blockIdx.x * 128;

    const int rr  = tid >> 1;           // tile row handled by this thread
    const int kk0 = (tid & 1) * 16;     // 16-element K half

    // Stage one 128x32 K-tile pair into LDS buffer `buf`.
    auto stage = [&](int k0, int buf) {
        if (!TA) {
            const unsigned l = (unsigned)(size_t)&As[buf][rr][kk0];
            const unsigned go = (unsigned)(((long)(m0 + rr) * lda + (k0 + kk0)) * 2);
            ASYNC_LDS_B128(l,       abase, go);
            ASYNC_LDS_B128(l + 16u, abase, go + 16u);
        } else {
            bf16 t[16];
            #pragma unroll
            for (int j = 0; j < 16; ++j)
                t[j] = Ab[(long)(k0 + kk0 + j) * lda + (m0 + rr)];
            v8bf lo, hi;
            #pragma unroll
            for (int j = 0; j < 8; ++j) { lo[j] = t[j]; hi[j] = t[8 + j]; }
            *(v8bf*)&As[buf][rr][kk0]     = lo;
            *(v8bf*)&As[buf][rr][kk0 + 8] = hi;
        }
        if (!TB) {
            const unsigned l = (unsigned)(size_t)&Bs[buf][rr][kk0];
            const unsigned go = (unsigned)(((long)(n0 + rr) * ldb + (k0 + kk0)) * 2);
            ASYNC_LDS_B128(l,       bbase, go);
            ASYNC_LDS_B128(l + 16u, bbase, go + 16u);
        } else {
            v8bf lo = {}, hi = {};
            if (n0 + rr < Nc) {
                bf16 t[16];
                #pragma unroll
                for (int j = 0; j < 16; ++j)
                    t[j] = Bb[(long)(k0 + kk0 + j) * ldb + (n0 + rr)];
                #pragma unroll
                for (int j = 0; j < 8; ++j) { lo[j] = t[j]; hi[j] = t[8 + j]; }
            }
            *(v8bf*)&Bs[buf][rr][kk0]     = lo;
            *(v8bf*)&Bs[buf][rr][kk0 + 8] = hi;
        }
    };

    v8f acc[4][2];
    #pragma unroll
    for (int i = 0; i < 4; ++i)
        #pragma unroll
        for (int j = 0; j < 2; ++j)
            acc[i][j] = {};

    stage(0, 0);
    int buf = 0;
    for (int k0 = 0; k0 < K; k0 += 32) {
        const bool has_next = (k0 + 32) < K;
        if (has_next) stage(k0 + 32, buf ^ 1);   // prefetch next tile
        if (PTILE > 0) {
            // wait until only the just-issued next-tile asyncs remain outstanding
            if (has_next) {
                if (PTILE == 4) asm volatile("s_wait_asynccnt 0x4" ::: "memory");
                else            asm volatile("s_wait_asynccnt 0x2" ::: "memory");
            } else {
                asm volatile("s_wait_asynccnt 0x0" ::: "memory");
            }
        }
        __syncthreads();

        // ---- build fragments (ds_read_b128 pairs) ----
        v16bf af[4], bfr[2];
        #pragma unroll
        for (int ti = 0; ti < 4; ++ti) {
            const int r  = wM * 64 + ti * 16 + (lane & 15);
            const int kb = (lane >> 4) * 8;     // half-wave K base
            v8bf lo = *(const v8bf*)&As[buf][r][kb];
            v8bf hi = *(const v8bf*)&As[buf][r][16 + kb];
            af[ti] = __builtin_shufflevector(lo, hi,
                      0,1,2,3,4,5,6,7,8,9,10,11,12,13,14,15);
        }
        #pragma unroll
        for (int tj = 0; tj < 2; ++tj) {
            const int c  = wN * 32 + tj * 16 + (lane & 15);
            const int ks = (lane >> 4) * 8;
            v8bf lo = *(const v8bf*)&Bs[buf][c][ks];
            v8bf hi = *(const v8bf*)&Bs[buf][c][ks + 16];
            bfr[tj] = __builtin_shufflevector(lo, hi,
                      0,1,2,3,4,5,6,7,8,9,10,11,12,13,14,15);
        }

        // ---- 8 WMMAs per wave per K-step ----
        #pragma unroll
        for (int ti = 0; ti < 4; ++ti)
            #pragma unroll
            for (int tj = 0; tj < 2; ++tj)
                acc[ti][tj] = __builtin_amdgcn_wmma_f32_16x16x32_bf16(
                    false, af[ti], false, bfr[tj], (short)0, acc[ti][tj],
                    false, false);
        __syncthreads();
        buf ^= 1;
    }

    // ---- epilogue: C/D layout is row = r + 8*(lane>=16), col = lane&15 ----
    // Hoisted bias (2 loads/wave); only wave-uniform column-tile guards.
    float bcol[2] = {0.f, 0.f};
    if (MODE == EP_F32 || MODE == EP_QKV) {
        #pragma unroll
        for (int tj = 0; tj < 2; ++tj)
            if (bias) bcol[tj] = bias[n0 + wN * 32 + tj * 16 + (lane & 15)];
    }
    #pragma unroll
    for (int ti = 0; ti < 4; ++ti) {
        #pragma unroll
        for (int tj = 0; tj < 2; ++tj) {
            const int colbase = n0 + wN * 32 + tj * 16;
            if (colbase >= Nc) continue;               // uniform branch
            const int col = colbase + (lane & 15);
            #pragma unroll
            for (int r = 0; r < 8; ++r) {
                const int row = m0 + wM * 64 + ti * 16 + r + ((lane >> 4) << 3);
                float v = acc[ti][tj][r];
                if (MODE == EP_F32) {
                    Cf[(long)row * ldc + col] = v + bcol[tj];
                } else if (MODE == EP_QKV) {
                    v = (v + bcol[tj]) * scale;
                    const int b = row >> 12, nl = row & 4095;
                    const int h = col >> 6,  d  = col & 63;
                    Cb[((long)((b * Ht + h) * Nt + nl)) * Dt + d] = (bf16)v;
                } else if (MODE == EP_BF16) {
                    Cb[(long)z * (long)Mr * ldc + (long)row * ldc + col] = (bf16)v;
                } else { // EP_OUT: scale row by d_inv, write merged [B,N,E]
                    v *= dinv[(long)z * Nt + row];
                    const int b = z >> 4, h = z & 15;
                    Cb[((long)(b * Nt + row)) * Et + h * Dt + col] = (bf16)v;
                }
            }
        }
    }
}

// ---------------------------------------------------------------------------
// Vectorized f32 -> bf16 conversion (8 elements per step)
// ---------------------------------------------------------------------------
__global__ __launch_bounds__(256)
void cvt_bf16(const float* __restrict__ s, bf16* __restrict__ d, long n8)
{
    long i = (long)blockIdx.x * blockDim.x + threadIdx.x;
    const long st = (long)gridDim.x * blockDim.x;
    for (; i < n8; i += st) {
        v4f a = ((const v4f*)s)[2 * i];
        v4f b = ((const v4f*)s)[2 * i + 1];
        v8bf o;
        #pragma unroll
        for (int j = 0; j < 4; ++j) { o[j] = (bf16)a[j]; o[4 + j] = (bf16)b[j]; }
        ((v8bf*)d)[i] = o;
    }
}

// ---------------------------------------------------------------------------
// Order-preserving float<->uint encoding for atomicMax on floats
// ---------------------------------------------------------------------------
__device__ inline unsigned enc_f(float f) {
    unsigned u = __float_as_uint(f);
    return (u & 0x80000000u) ? ~u : (u | 0x80000000u);
}
__device__ inline float dec_f(unsigned k) {
    return (k & 0x80000000u) ? __uint_as_float(k ^ 0x80000000u)
                             : __uint_as_float(~k);
}

__global__ void init_gmax(unsigned* g) { if (threadIdx.x == 0 && blockIdx.x == 0) *g = 0u; }

__global__ __launch_bounds__(256)
void reduce_max_k(const float* __restrict__ d, unsigned* __restrict__ g, long n)
{
    long i = (long)blockIdx.x * blockDim.x + threadIdx.x;
    const long st = (long)gridDim.x * blockDim.x;
    float mx = -3.4e38f;
    for (; i < n; i += st) mx = fmaxf(mx, d[i]);
    #pragma unroll
    for (int o = 16; o; o >>= 1) mx = fmaxf(mx, __shfl_xor(mx, o, 32));
    __shared__ float smx[8];
    if ((threadIdx.x & 31) == 0) smx[threadIdx.x >> 5] = mx;
    __syncthreads();
    if (threadIdx.x == 0) {
        float m2 = smx[0];
        #pragma unroll
        for (int w = 1; w < 8; ++w) m2 = fmaxf(m2, smx[w]);
        atomicMax(g, enc_f(m2));
    }
}

// ---------------------------------------------------------------------------
// FAVOR+ feature maps: one wave per (b,h,n) row; diag = 0.5*||data_n||^2
// ---------------------------------------------------------------------------
__global__ __launch_bounds__(256)
void feat_q(const float* __restrict__ dd, const bf16* __restrict__ qs,
            bf16* __restrict__ qf)
{
    const int row  = blockIdx.x * 8 + (threadIdx.x >> 5);
    const int lane = threadIdx.x & 31;
    const bf16* qr = qs + (long)row * Dt;
    float sq = 0.f;
    #pragma unroll
    for (int j = 0; j < 2; ++j) { float v = (float)qr[lane * 2 + j]; sq += v * v; }
    #pragma unroll
    for (int o = 16; o; o >>= 1) sq += __shfl_xor(sq, o, 32);
    const float diag = 0.5f * sq;

    const float* dr = dd + (long)row * Mt;
    float v[8]; float mx = -3.4e38f;
    #pragma unroll
    for (int j = 0; j < 8; ++j) { v[j] = dr[lane * 8 + j]; mx = fmaxf(mx, v[j]); }
    #pragma unroll
    for (int o = 16; o; o >>= 1) mx = fmaxf(mx, __shfl_xor(mx, o, 32));

    bf16* out = qf + (long)row * Mt;
    #pragma unroll
    for (int j = 0; j < 8; ++j)
        out[lane * 8 + j] = (bf16)(RATIO * (__expf(v[j] - diag - mx) + KEPS));
}

__global__ __launch_bounds__(256)
void feat_k(const float* __restrict__ dd, const bf16* __restrict__ ks,
            bf16* __restrict__ kf, const unsigned* __restrict__ gmax)
{
    const int row  = blockIdx.x * 8 + (threadIdx.x >> 5);
    const int lane = threadIdx.x & 31;
    const bf16* kr = ks + (long)row * Dt;
    float sq = 0.f;
    #pragma unroll
    for (int j = 0; j < 2; ++j) { float v = (float)kr[lane * 2 + j]; sq += v * v; }
    #pragma unroll
    for (int o = 16; o; o >>= 1) sq += __shfl_xor(sq, o, 32);
    const float diag = 0.5f * sq;
    const float mx   = dec_f(*gmax);

    const float* dr = dd + (long)row * Mt;
    bf16* out = kf + (long)row * Mt;
    #pragma unroll
    for (int j = 0; j < 8; ++j) {
        float v = dr[lane * 8 + j];
        out[lane * 8 + j] = (bf16)(RATIO * (__expf(v - diag - mx) + KEPS));
    }
}

// k_sum[b,h,m] = sum_n kf[b,h,n,m]   (one block per (b,h), threads = m)
__global__ __launch_bounds__(256)
void ksum_k(const bf16* __restrict__ kf, float* __restrict__ ksum)
{
    const int z = blockIdx.x, m = threadIdx.x;
    const bf16* p = kf + (long)z * Nt * Mt + m;
    float acc = 0.f;
    for (int n = 0; n < Nt; ++n) acc += (float)p[(long)n * Mt];
    ksum[z * Mt + m] = acc;
}

// d_inv[row] = 1 / dot(qf[row,:], k_sum[z,:])   (one wave per row)
__global__ __launch_bounds__(256)
void dinv_k(const bf16* __restrict__ qf, const float* __restrict__ ksum,
            float* __restrict__ dinv)
{
    const int row  = blockIdx.x * 8 + (threadIdx.x >> 5);
    const int lane = threadIdx.x & 31;
    const int z    = row >> 12;
    const bf16*  q = qf + (long)row * Mt;
    const float* s = ksum + z * Mt;
    float acc = 0.f;
    #pragma unroll
    for (int j = 0; j < 8; ++j) { int m = lane * 8 + j; acc += (float)q[m] * s[m]; }
    #pragma unroll
    for (int o = 16; o; o >>= 1) acc += __shfl_xor(acc, o, 32);
    if (lane == 0) dinv[row] = 1.f / acc;
}

// ---------------------------------------------------------------------------
extern "C" void kernel_launch(void* const* d_in, const int* in_sizes, int n_in,
                              void* d_out, int out_size, void* d_ws, size_t ws_size,
                              hipStream_t stream)
{
    (void)in_sizes; (void)n_in; (void)out_size; (void)ws_size;
    const float* x    = (const float*)d_in[0];
    const float* Wq   = (const float*)d_in[1];
    const float* bq   = (const float*)d_in[2];
    const float* Wk   = (const float*)d_in[3];
    const float* bk   = (const float*)d_in[4];
    const float* Wv   = (const float*)d_in[5];
    const float* bv   = (const float*)d_in[6];
    const float* Wo   = (const float*)d_in[7];
    const float* bo   = (const float*)d_in[8];
    const float* proj = (const float*)d_in[9];
    float* out = (float*)d_out;

    // Workspace carve-up (bytes)
    char* ws = (char*)d_ws;
    size_t off = 0;
    auto carve = [&](size_t bytes) { void* p = ws + off; off += (bytes + 255) & ~(size_t)255; return p; };
    bf16*     xb    = (bf16*)carve((size_t)ROWS * Et * 2);     // x in bf16
    bf16*     wqb   = (bf16*)carve((size_t)Et * Et * 2);
    bf16*     wkb   = (bf16*)carve((size_t)Et * Et * 2);
    bf16*     wvb   = (bf16*)carve((size_t)Et * Et * 2);
    bf16*     wob   = (bf16*)carve((size_t)Et * Et * 2);
    bf16*     projb = (bf16*)carve((size_t)Mt * Dt * 2);
    bf16*     qs    = (bf16*)carve((size_t)FR * Dt * 2);       // head-split, scaled
    bf16*     kk    = (bf16*)carve((size_t)FR * Dt * 2);
    bf16*     vs    = (bf16*)carve((size_t)FR * Dt * 2);
    float*    dd    = (float*)carve((size_t)FR * Mt * 4);      // reused for q then k
    bf16*     qf    = (bf16*)carve((size_t)FR * Mt * 2);
    bf16*     kf    = (bf16*)carve((size_t)FR * Mt * 2);
    bf16*     ctx   = (bf16*)carve((size_t)Bt * Ht * Mt * Dt * 2);
    float*    ksum  = (float*)carve((size_t)Bt * Ht * Mt * 4);
    float*    dinv  = (float*)carve((size_t)FR * 4);
    bf16*     attn  = (bf16*)carve((size_t)ROWS * Et * 2);     // merged [B,N,E]
    unsigned* gmax  = (unsigned*)carve(256);

    const dim3 blk(256);

    // 1) convert inputs to bf16 (8 elems/thread-step)
    cvt_bf16<<<1024, blk, 0, stream>>>(x,    xb,    (long)ROWS * Et / 8);
    cvt_bf16<<<256,  blk, 0, stream>>>(Wq,   wqb,   (long)Et * Et / 8);
    cvt_bf16<<<256,  blk, 0, stream>>>(Wk,   wkb,   (long)Et * Et / 8);
    cvt_bf16<<<256,  blk, 0, stream>>>(Wv,   wvb,   (long)Et * Et / 8);
    cvt_bf16<<<256,  blk, 0, stream>>>(Wo,   wob,   (long)Et * Et / 8);
    cvt_bf16<<<8,    blk, 0, stream>>>(proj, projb, (long)Mt * Dt / 8);

    // 2) QKV projections -> head-split bf16 (q,k pre-scaled by d^-0.25)
    {
        dim3 g(Et / 128, ROWS / 128, 1);
        wmma_gemm<EP_QKV, false, false><<<g, blk, 0, stream>>>(
            xb, wqb, bq, nullptr, nullptr, qs, ROWS, Et, Et, Et, Et, Et, 0, 0, QK_SCALE);
        wmma_gemm<EP_QKV, false, false><<<g, blk, 0, stream>>>(
            xb, wkb, bk, nullptr, nullptr, kk, ROWS, Et, Et, Et, Et, Et, 0, 0, QK_SCALE);
        wmma_gemm<EP_QKV, false, false><<<g, blk, 0, stream>>>(
            xb, wvb, bv, nullptr, nullptr, vs, ROWS, Et, Et, Et, Et, Et, 0, 0, 1.0f);
    }

    // 3) dd_q = qn @ proj^T  -> f32 ; then qf features (per-row max stab)
    {
        dim3 g(Mt / 128, FR / 128, 1);
        wmma_gemm<EP_F32, false, false><<<g, blk, 0, stream>>>(
            qs, projb, nullptr, nullptr, dd, nullptr, FR, Mt, Dt, Dt, Dt, Mt, 0, 0, 1.0f);
    }
    feat_q<<<FR / 8, blk, 0, stream>>>(dd, qs, qf);

    // 4) dd_k ; global max stabilizer ; kf features
    {
        dim3 g(Mt / 128, FR / 128, 1);
        wmma_gemm<EP_F32, false, false><<<g, blk, 0, stream>>>(
            kk, projb, nullptr, nullptr, dd, nullptr, FR, Mt, Dt, Dt, Dt, Mt, 0, 0, 1.0f);
    }
    init_gmax<<<1, 1, 0, stream>>>(gmax);
    reduce_max_k<<<2048, blk, 0, stream>>>(dd, gmax, (long)FR * Mt);
    feat_k<<<FR / 8, blk, 0, stream>>>(dd, kk, kf, gmax);

    // 5) k_sum and context = kf^T @ v   (batched over B*H, A transposed, B "transposed")
    ksum_k<<<Bt * Ht, blk, 0, stream>>>(kf, ksum);
    {
        dim3 g(1, Mt / 128, Bt * Ht);
        wmma_gemm<EP_BF16, true, true><<<g, blk, 0, stream>>>(
            kf, vs, nullptr, nullptr, nullptr, ctx, Mt, Dt, Nt,
            Mt /*lda: kf row stride*/, Dt /*ldb: v row stride*/, Dt,
            (long)Nt * Mt, (long)Nt * Dt, 1.0f);
    }

    // 6) d_inv and out = d_inv * (qf @ context)  -> merged bf16 [B,N,E]
    dinv_k<<<FR / 8, blk, 0, stream>>>(qf, ksum, dinv);
    {
        dim3 g(1, Nt / 128, Bt * Ht);
        wmma_gemm<EP_OUT, false, true><<<g, blk, 0, stream>>>(
            qf, ctx, nullptr, dinv, nullptr, attn, Nt, Dt, Mt,
            Mt /*lda*/, Dt /*ldb*/, Dt,
            (long)Nt * Mt, (long)Mt * Dt, 1.0f);
    }

    // 7) final projection: out = attn @ Wo^T + bo  -> f32 d_out
    {
        dim3 g(Et / 128, ROWS / 128, 1);
        wmma_gemm<EP_F32, false, false><<<g, blk, 0, stream>>>(
            attn, wob, bo, nullptr, out, nullptr, ROWS, Et, Et, Et, Et, Et, 0, 0, 1.0f);
    }
}
